// MaskedMultiHeadAttn_26070451486893
// MI455X (gfx1250) — compile-verified, matrix operand order checked by reading, not run on hardware
//
#include <hip/hip_runtime.h>
#include <cstdint>

#define B_  4
#define S_  1024
#define H_  1024
#define NH_ 16
#define HS_ 64
#define M_  (B_*S_)   // 4096 rows total

typedef __attribute__((ext_vector_type(16))) __bf16 v16bf;
typedef __attribute__((ext_vector_type(2)))  __bf16 bf16x2;
typedef __attribute__((ext_vector_type(8)))  float  v8f;
typedef __attribute__((ext_vector_type(4)))  unsigned int u32x4;
typedef __attribute__((ext_vector_type(8)))  int          i32x8;
typedef __attribute__((ext_vector_type(4)))  int          i32x4;

#if defined(__has_builtin)
#if __has_builtin(__builtin_amdgcn_tensor_load_to_lds) && __has_builtin(__builtin_amdgcn_s_wait_tensorcnt)
#define USE_TDM 1
#endif
#endif
#ifndef USE_TDM
#define USE_TDM 0
#endif

__device__ __forceinline__ unsigned short f32_to_bf16(float f){
  unsigned int u = __builtin_bit_cast(unsigned int, f);
  u += 0x7FFFu + ((u >> 16) & 1u);            // round-to-nearest-even
  return (unsigned short)(u >> 16);
}
__device__ __forceinline__ float bf16_to_f32(unsigned short h){
  unsigned int u = ((unsigned int)h) << 16;
  return __builtin_bit_cast(float, u);
}
__device__ __forceinline__ v8f vzero8(){
  v8f z;
  #pragma unroll
  for (int i=0;i<8;++i) z[i] = 0.0f;
  return z;
}

#if USE_TDM
// Tensor Data Mover: load a [tile_d1 x tile_d0] bf16 tile (row stride stride_elems
// in global) into LDS with a 1-DWORD pad inserted every 2^(pad_code+1) DWORDs
// (matches padded LDS row strides like 66 or 34 ushorts).
// D# layout per CDNA5 ISA ch.8: group0 = {flags, lds_addr, global_addr_lo, global_addr_hi|type},
// group1 = {mask/size/pad, tensor dims, tile dims, dim0 stride}.
__device__ __forceinline__ void tdm_load_tile_bf16(unsigned int lds_off, const void* gptr,
                                                   int tile_d0, int tile_d1,
                                                   int tensor_d0, int tensor_d1,
                                                   int stride_elems, int pad_code)
{
  unsigned long long ga = (unsigned long long)(uintptr_t)gptr;
  u32x4 g0 = { 1u,                                   // count=1, user descriptor
               lds_off,                              // LDS byte address
               (unsigned int)ga,                     // global addr [31:0]
               ((unsigned int)(ga >> 32) & 0x01FFFFFFu) | (2u << 30) }; // [56:32] + type=2
  i32x8 g1 = {
    (int)((1u << 16) | (1u << 20) | ((unsigned)pad_code << 22)), // data_size=2B, pad_enable, pad_interval; pad_amount=0 (1 DWORD)
    (int)(((unsigned)tensor_d0 & 0xFFFFu) << 16),                // tensor_dim0[15:0] at [31:16]
    (int)((((unsigned)tensor_d0 >> 16) & 0xFFFFu) | (((unsigned)tensor_d1 & 0xFFFFu) << 16)),
    (int)((((unsigned)tensor_d1 >> 16) & 0xFFFFu) | (((unsigned)tile_d0 & 0xFFFFu) << 16)),
    (int)((unsigned)tile_d1 & 0xFFFFu),                          // tile_dim1; tile_dim2=0
    stride_elems,                                                // tensor_dim0_stride[31:0]
    0, 0 };
  i32x4 z4 = {0,0,0,0};
#if __clang_major__ >= 23
  i32x8 z8 = {0,0,0,0,0,0,0,0};
  __builtin_amdgcn_tensor_load_to_lds(g0, g1, z4, z4, z8, 0);
#else
  __builtin_amdgcn_tensor_load_to_lds(g0, g1, z4, z4, 0);
#endif
}
#endif

// A fragment (16 rows x 32 K, bf16) from a row-major ushort buffer.
// ISA layout: lanes 0-15 row M=lane, K in {0-7,16-23}; lanes 16-31 same rows, K in {8-15,24-31}.
__device__ __forceinline__ v16bf load_frag_a(const unsigned short* base, int ld,
                                             int row0, int k0, int lane){
  v16bf a;
  const unsigned short* p = base + (size_t)(row0 + (lane & 15)) * ld + k0 + ((lane >> 4) << 3);
  #pragma unroll
  for (int v = 0; v < 8; ++v){
    int kk = (v < 4) ? (v * 2) : (16 + (v - 4) * 2);
    bf16x2 t = __builtin_bit_cast(bf16x2, *(const unsigned int*)(p + kk));
    a[2*v] = t.x; a[2*v+1] = t.y;
  }
  return a;
}

// B fragment (32 K x 16 N, bf16) from a row-major [k][n] ushort buffer (k0 folded into base).
__device__ __forceinline__ v16bf load_frag_b(const unsigned short* base, int ld,
                                             int col0, int lane){
  v16bf b;
  int c  = col0 + (lane & 15);
  int kb = (lane >> 4) << 3;
  #pragma unroll
  for (int v = 0; v < 8; ++v){
    int kk = ((v < 4) ? (v*2) : (16 + (v-4)*2)) + kb;
    b[2*v]   = __builtin_bit_cast(__bf16, base[(size_t)kk     * ld + c]);
    b[2*v+1] = __builtin_bit_cast(__bf16, base[(size_t)(kk+1) * ld + c]);
  }
  return b;
}

__device__ __forceinline__ v8f wmma_bf16(v16bf a, v16bf b, v8f c){
  return __builtin_amdgcn_wmma_f32_16x16x32_bf16(false, a, false, b, (short)0, c, false, false);
}

// ---------------------------------------------------------------------------
// Tiled WMMA GEMM: C[M,N] = A[M,K] @ W[K,N] + bias   (W is f32, cast to bf16)
// ---------------------------------------------------------------------------
template<bool A_IS_F32, bool OUT_IS_F32>
__global__ __launch_bounds__(256)
void gemm_wmma(const void* __restrict__ Ap, const float* __restrict__ W,
               const float* __restrict__ bias, void* __restrict__ Cp,
               int Mdim, int Ndim, int Kdim)
{
  constexpr int BM = 128, BN = 128, BK = 32;
  __shared__ unsigned short As[BM][BK + 2];
  __shared__ unsigned short Bs[BK][BN + 2];

  const int tid  = threadIdx.x;
  const int lane = tid & 31;
  const int wave = tid >> 5;
  const int wm   = wave >> 1;          // 4 waves down: 32-row strip each
  const int wn   = wave & 1;           // 2 waves across: 64-col strip each
  const int m0   = blockIdx.y * BM;
  const int n0   = blockIdx.x * BN;

  const float*          Af = (const float*)Ap;
  const unsigned short* Ab = (const unsigned short*)Ap;

  v8f acc[2][4];
  #pragma unroll
  for (int i=0;i<2;++i)
    #pragma unroll
    for (int j=0;j<4;++j) acc[i][j] = vzero8();

  for (int kt = 0; kt < Kdim; kt += BK){
    // stage A tile (bf16 source: TDM DMA; f32 source: VALU convert + ds_store)
    bool staged_by_tdm = false;
#if USE_TDM
    if constexpr (!A_IS_F32){
      if (wave == 0){
        tdm_load_tile_bf16((unsigned int)(uintptr_t)&As[0][0],
                           Ab + (size_t)m0*Kdim + kt,
                           /*tile*/ BK, BM, /*tensor*/ Kdim, Mdim,
                           /*stride*/ Kdim, /*pad: 16 DWORDs*/ 3);
        __builtin_amdgcn_s_wait_tensorcnt(0);
      }
      staged_by_tdm = true;
    }
#endif
    if (!staged_by_tdm){
      #pragma unroll
      for (int i = tid; i < BM*BK; i += 256){
        int r = i >> 5, c = i & 31;
        unsigned short hv;
        if (A_IS_F32) hv = f32_to_bf16(Af[(size_t)(m0+r)*Kdim + kt + c]);
        else          hv = Ab[(size_t)(m0+r)*Kdim + kt + c];
        As[r][c] = hv;
      }
    }
    // stage W tile (f32 -> bf16 LDS)
    #pragma unroll
    for (int i = tid; i < BK*BN; i += 256){
      int r = i >> 7, c = i & 127;
      Bs[r][c] = f32_to_bf16(W[(size_t)(kt + r)*Ndim + n0 + c]);
    }
    // hint next W tile into cache (gfx1250 global_prefetch_b8)
    if (kt + BK < Kdim)
      __builtin_prefetch(&W[(size_t)(kt + BK)*Ndim + n0 + (tid & 127)], 0, 0);
    __syncthreads();

    v16bf afr[2], bfr[4];
    #pragma unroll
    for (int i=0;i<2;++i) afr[i] = load_frag_a(&As[0][0], BK+2, wm*32 + i*16, 0, lane);
    #pragma unroll
    for (int j=0;j<4;++j) bfr[j] = load_frag_b(&Bs[0][0], BN+2, wn*64 + j*16, lane);
    #pragma unroll
    for (int i=0;i<2;++i)
      #pragma unroll
      for (int j=0;j<4;++j)
        acc[i][j] = wmma_bf16(afr[i], bfr[j], acc[i][j]);
    __syncthreads();
  }

  // epilogue: + bias, store (C layout: VGPR v -> M=v (lanes 0-15) / M=8+v (lanes 16-31))
  float*          Cf = (float*)Cp;
  unsigned short* Cb = (unsigned short*)Cp;
  const int half = (lane >> 4) << 3;
  #pragma unroll
  for (int i=0;i<2;++i){
    #pragma unroll
    for (int j=0;j<4;++j){
      int col = n0 + wn*64 + j*16 + (lane & 15);
      float bv = bias[col];
      #pragma unroll
      for (int v=0;v<8;++v){
        int row = m0 + wm*32 + i*16 + v + half;
        float val = acc[i][j][v] + bv;
        if (OUT_IS_F32) Cf[(size_t)row*Ndim + col] = val;
        else            Cb[(size_t)row*Ndim + col] = f32_to_bf16(val);
      }
    }
  }
}

// ---------------------------------------------------------------------------
// RoPE: out[j] = x0*cos - x1*sin ; out[j+32] = x1*cos + x0*sin  (per 64-d head)
// blockIdx.y: 0 -> Q with glycan_query_mass, 1 -> K with glycan_mass
// ---------------------------------------------------------------------------
__global__ __launch_bounds__(256)
void rope_kernel(const unsigned short* __restrict__ Qb, const unsigned short* __restrict__ Kb,
                 const float* __restrict__ gqm, const float* __restrict__ gm,
                 unsigned short* __restrict__ Qr, unsigned short* __restrict__ Kr)
{
  int idx = blockIdx.x * 256 + threadIdx.x;
  const unsigned short* src = blockIdx.y ? Kb : Qb;
  unsigned short*       dst = blockIdx.y ? Kr : Qr;
  const float*          dis = blockIdx.y ? gm : gqm;

  int j    = idx & 31;
  int head = (idx >> 5) & 15;
  int s    = (idx >> 9) & 1023;
  int b    = idx >> 19;

  size_t dbase = ((size_t)b * S_ + s) * HS_;
  float sv = dis[dbase + j];
  float cv = dis[dbase + 32 + j];
  size_t base = ((size_t)b * S_ + s) * H_ + head * HS_;
  float x0 = bf16_to_f32(src[base + 2*j]);
  float x1 = bf16_to_f32(src[base + 2*j + 1]);
  dst[base + j]      = f32_to_bf16(x0 * cv - x1 * sv);
  dst[base + 32 + j] = f32_to_bf16(x1 * cv + x0 * sv);
}

// ---------------------------------------------------------------------------
// Flash-style causal attention, one block per (b, head, 64-row q tile),
// 4 waves, each owning a 16-row strip. All matmuls via bf16 WMMA.
// K/V tiles staged by the Tensor Data Mover when available.
// ---------------------------------------------------------------------------
__global__ __launch_bounds__(128)
void attn_kernel(const unsigned short* __restrict__ Qr,
                 const unsigned short* __restrict__ Kr,
                 const unsigned short* __restrict__ Vb,
                 unsigned short* __restrict__ AO)
{
  __shared__ unsigned short Kt[64][66];
  __shared__ unsigned short Vt[64][66];
  __shared__ unsigned short Ps[64][66];

  const int tid  = threadIdx.x;
  const int lane = tid & 31;
  const int wave = tid >> 5;                 // 0..3
  const int m0   = blockIdx.x * 64;
  const int head = blockIdx.y;
  const int b    = blockIdx.z;
  const int half = (lane >> 4) << 3;
  const size_t bS = (size_t)b * S_;

  // Q fragments (16 rows x 64 d) gathered directly from global, per ISA A layout
  const unsigned short* qp = Qr + (bS + m0 + wave*16 + (lane & 15)) * H_ + head * HS_;
  v16bf qf[2];
  #pragma unroll
  for (int c = 0; c < 2; ++c){
    v16bf a;
    const unsigned short* p = qp + c*32 + half;
    #pragma unroll
    for (int v = 0; v < 8; ++v){
      int kk = (v < 4) ? (v*2) : (16 + (v-4)*2);
      bf16x2 t = __builtin_bit_cast(bf16x2, *(const unsigned int*)(p + kk));
      a[2*v] = t.x; a[2*v+1] = t.y;
    }
    qf[c] = a;
  }

  v8f oacc[4];
  #pragma unroll
  for (int h=0;h<4;++h) oacc[h] = vzero8();
  float rmax[8], rsum[8];
  #pragma unroll
  for (int v=0;v<8;++v){ rmax[v] = -1e30f; rsum[v] = 0.0f; }

  const int ntiles = blockIdx.x + 1;           // causal: only tiles up to the diagonal
  for (int jt = 0; jt < ntiles; ++jt){
    const int kv0 = jt * 64;
    // stage K,V tiles into padded LDS
#if USE_TDM
    if (wave == 0){
      tdm_load_tile_bf16((unsigned int)(uintptr_t)&Kt[0][0],
                         Kr + (bS + kv0) * H_ + head * HS_,
                         /*tile*/ HS_, 64, /*tensor*/ H_, M_,
                         /*stride*/ H_, /*pad: 32 DWORDs*/ 4);
      tdm_load_tile_bf16((unsigned int)(uintptr_t)&Vt[0][0],
                         Vb + (bS + kv0) * H_ + head * HS_,
                         HS_, 64, H_, M_, H_, 4);
      __builtin_amdgcn_s_wait_tensorcnt(0);
    }
#else
    for (int i = tid; i < 64*32; i += 128){
      int r = i >> 5, c2 = (i & 31) * 2;
      size_t g = (bS + kv0 + r) * H_ + head * HS_ + c2;
      *(unsigned int*)&Kt[r][c2] = *(const unsigned int*)&Kr[g];
      *(unsigned int*)&Vt[r][c2] = *(const unsigned int*)&Vb[g];
    }
#endif
    __syncthreads();

    // S strip (16 x 64) = Q_strip @ K^T  (B-frag = transposed gather from Kt)
    v8f sacc[4];
    #pragma unroll
    for (int t=0;t<4;++t) sacc[t] = vzero8();
    #pragma unroll
    for (int t=0;t<4;++t)
      #pragma unroll
      for (int c=0;c<2;++c)
        sacc[t] = wmma_bf16(qf[c], load_frag_a(&Kt[0][0], 66, t*16, c*32, lane), sacc[t]);

    // scale, causal mask, per-row max (rows live in 16-lane groups)
    float tmax[8];
    #pragma unroll
    for (int v=0;v<8;++v) tmax[v] = -1e30f;
    const int qrbase = m0 + wave*16 + half;
    #pragma unroll
    for (int t=0;t<4;++t){
      int kc = kv0 + t*16 + (lane & 15);
      #pragma unroll
      for (int v=0;v<8;++v){
        float sv = sacc[t][v] * 0.125f;           // 1/sqrt(64)
        if (kc > qrbase + v) sv = -1e30f;
        sacc[t][v] = sv;
        tmax[v] = fmaxf(tmax[v], sv);
      }
    }
    #pragma unroll
    for (int v=0;v<8;++v)
      #pragma unroll
      for (int msk=1; msk<16; msk<<=1)
        tmax[v] = fmaxf(tmax[v], __shfl_xor(tmax[v], msk, 32));

    // online softmax update
    float psum[8];
    #pragma unroll
    for (int v=0;v<8;++v){
      float mnew = fmaxf(rmax[v], tmax[v]);
      float sc   = __expf(rmax[v] - mnew);
      rmax[v] = mnew;
      rsum[v] *= sc;
      #pragma unroll
      for (int h=0;h<4;++h) oacc[h][v] *= sc;
      psum[v] = 0.0f;
    }
    #pragma unroll
    for (int t=0;t<4;++t){
      #pragma unroll
      for (int v=0;v<8;++v){
        float p = __expf(sacc[t][v] - rmax[v]);
        psum[v] += p;
        Ps[wave*16 + v + half][t*16 + (lane & 15)] = f32_to_bf16(p);
      }
    }
    #pragma unroll
    for (int v=0;v<8;++v){
      #pragma unroll
      for (int msk=1; msk<16; msk<<=1)
        psum[v] += __shfl_xor(psum[v], msk, 32);
      rsum[v] += psum[v];
    }
    __syncthreads();

    // O_strip += P_strip @ V
    #pragma unroll
    for (int h=0;h<4;++h)
      #pragma unroll
      for (int c=0;c<2;++c)
        oacc[h] = wmma_bf16(load_frag_a(&Ps[0][0], 66, wave*16, c*32, lane),
                            load_frag_b(&Vt[c*32][0], 66, h*16, lane),
                            oacc[h]);
    __syncthreads();
  }

  // normalize and store (bf16, [b,s,h] layout feeding the Wo GEMM)
  #pragma unroll
  for (int h=0;h<4;++h){
    int col = head * HS_ + h*16 + (lane & 15);
    #pragma unroll
    for (int v=0;v<8;++v){
      int qr = m0 + wave*16 + v + half;
      AO[(bS + qr) * H_ + col] = f32_to_bf16(oacc[h][v] / rsum[v]);
    }
  }
}

// ---------------------------------------------------------------------------
// x = tgt*ALPHA + proj ; LayerNorm over H. One block (256 threads) per row.
// ---------------------------------------------------------------------------
__global__ __launch_bounds__(256)
void resid_ln_kernel(const float* __restrict__ tgt, const float* __restrict__ proj,
                     const float* __restrict__ lnw, const float* __restrict__ lnb,
                     float* __restrict__ out)
{
  __shared__ float ssum[256], ssq[256];
  const int row = blockIdx.x;
  const int tid = threadIdx.x;
  float x[4], s = 0.0f, q = 0.0f;
  #pragma unroll
  for (int i=0;i<4;++i){
    int col = tid + i*256;
    float v = tgt[(size_t)row*H_ + col] * 2.0f + proj[(size_t)row*H_ + col];
    x[i] = v; s += v; q += v*v;
  }
  ssum[tid] = s; ssq[tid] = q;
  __syncthreads();
  for (int off = 128; off > 0; off >>= 1){
    if (tid < off){ ssum[tid] += ssum[tid+off]; ssq[tid] += ssq[tid+off]; }
    __syncthreads();
  }
  float mean = ssum[0] * (1.0f / H_);
  float var  = ssq[0]  * (1.0f / H_) - mean * mean;
  float rstd = rsqrtf(var + 1e-5f);
  #pragma unroll
  for (int i=0;i<4;++i){
    int col = tid + i*256;
    out[(size_t)row*H_ + col] = (x[i] - mean) * rstd * lnw[col] + lnb[col];
  }
}

// ---------------------------------------------------------------------------
extern "C" void kernel_launch(void* const* d_in, const int* in_sizes, int n_in,
                              void* d_out, int out_size, void* d_ws, size_t ws_size,
                              hipStream_t stream)
{
  (void)in_sizes; (void)n_in; (void)out_size; (void)ws_size;
  const float* tgt = (const float*)d_in[0];
  // d_in[1] = tgt_mask (bool) — causal mask computed analytically instead
  const float* gm  = (const float*)d_in[2];
  const float* gqm = (const float*)d_in[3];
  const float* Wq  = (const float*)d_in[4];
  const float* bq  = (const float*)d_in[5];
  const float* Wk  = (const float*)d_in[6];
  const float* bk  = (const float*)d_in[7];
  const float* Wv  = (const float*)d_in[8];
  const float* bv  = (const float*)d_in[9];
  const float* Wo  = (const float*)d_in[10];
  const float* bo  = (const float*)d_in[11];
  const float* lnw = (const float*)d_in[12];
  const float* lnb = (const float*)d_in[13];
  float* out = (float*)d_out;

  const size_t E = (size_t)M_ * H_;          // 4M elements
  unsigned short* Qb = (unsigned short*)d_ws;
  unsigned short* Kb = Qb + E;
  unsigned short* Vb = Kb + E;
  unsigned short* Qr = Vb + E;
  unsigned short* Kr = Qr + E;
  unsigned short* AO = Kr + E;
  float*          PJ = (float*)(AO + E);     // 48 MB bf16 + 16 MB f32 = 64 MB total

  dim3 gg(H_/128, M_/128);                   // 8 x 32 blocks per GEMM
  gemm_wmma<true,  false><<<gg, 256, 0, stream>>>(tgt, Wq, bq, Qb, M_, H_, H_);
  gemm_wmma<true,  false><<<gg, 256, 0, stream>>>(tgt, Wk, bk, Kb, M_, H_, H_);
  gemm_wmma<true,  false><<<gg, 256, 0, stream>>>(tgt, Wv, bv, Vb, M_, H_, H_);
  rope_kernel<<<dim3((B_*S_*NH_*HS_/2)/256, 2), 256, 0, stream>>>(Qb, Kb, gqm, gm, Qr, Kr);
  attn_kernel<<<dim3(S_/64, NH_, B_), 128, 0, stream>>>(Qr, Kr, Vb, AO);
  gemm_wmma<false, true ><<<gg, 256, 0, stream>>>(AO, Wo, bo, PJ, M_, H_, H_);
  resid_ln_kernel<<<M_, 256, 0, stream>>>(tgt, PJ, lnw, lnb, out);
}
